// Gemma3PrefillWrapper_39298950758490
// MI455X (gfx1250) — matrix operand order repaired and checked by reading, not compiled, not run on hardware
//
#include <hip/hip_runtime.h>
#include <hip/hip_bf16.h>

// ---------------- model constants ----------------
#define L_   2
#define D_   2048
#define H_   8
#define KV_  4
#define HD_  256
#define FF_  8192
#define V_   32768
#define T_   512
#define NREP_ 2
#define EPS_ 1e-6f
#define SCALE_ 0.0625f              // HD^-0.5
#define EMB_SCALE_ 45.254833995939045f  // sqrt(2048)

typedef __attribute__((ext_vector_type(16))) __bf16 bf16x16;
typedef __attribute__((ext_vector_type(4)))  __bf16 bf16x4;
typedef __attribute__((ext_vector_type(8)))  float  floatx8;
typedef __attribute__((ext_vector_type(4)))  float  floatx4;

// ---------------- generalized batched WMMA GEMM ----------------
// C[z][m][n] = sum_k A[z][m][k] * B[z][n][k]
// A element (m,k): A + (z/Abdiv)*Ab + m*Am + k          (k-stride 1)
// B element (n,k): B + (z/Bbdiv)*Bb + n*Bn + k*Bk
// C element (m,n): C + z*Cb + m*Cm + n
#define BM 128
#define BN 128
#define BK 64
#define LDST 72   // LDS row stride in halves (144B: 16B aligned, conflict-breaking)

__global__ __launch_bounds__(256) void wmma_gemm_bf16(
    const float* __restrict__ A, const float* __restrict__ B, float* __restrict__ C,
    int M, int N, int K,
    int Am, int Bn, int Bk, int Cm,
    long long Ab, int Abdiv, long long Bb, int Bbdiv, long long Cb)
{
    __shared__ __bf16 As[BM * LDST];
    __shared__ __bf16 Bs[BN * LDST];

    const int tid  = threadIdx.x;
    const int lane = tid & 31;
    const int wave = tid >> 5;
    const int wm   = wave >> 1;   // 0..3 : 32-row stripe
    const int wn   = wave & 1;    // 0..1 : 64-col stripe
    const int g    = lane >> 4;   // half-wave group
    const int lr   = lane & 15;

    const int m0 = blockIdx.y * BM;
    const int n0 = blockIdx.x * BN;
    const int z  = blockIdx.z;

    const float* Ap = A + (long long)(z / Abdiv) * Ab;
    const float* Bp = B + (long long)(z / Bbdiv) * Bb;
    float*       Cp = C + (long long)z * Cb;

    floatx8 acc[2][4];
#pragma unroll
    for (int mt = 0; mt < 2; ++mt)
#pragma unroll
        for (int nt = 0; nt < 4; ++nt)
#pragma unroll
            for (int j = 0; j < 8; ++j) acc[mt][nt][j] = 0.0f;

    for (int k0 = 0; k0 < K; k0 += BK) {
        const bool pf = (k0 + BK) < K;

        // ---- stage A tile (128 x 64) fp32 -> bf16, vectorized along k ----
#pragma unroll
        for (int i = 0; i < 8; ++i) {
            int q  = tid + i * 256;
            int mm = q >> 4, kq = (q & 15) * 4;
            const float* src = Ap + (long long)(m0 + mm) * Am + (k0 + kq);
            floatx4 v = *(const floatx4*)src;
            if (pf) __builtin_prefetch(src + BK, 0, 3);
            bf16x4 c;
            c[0] = (__bf16)v[0]; c[1] = (__bf16)v[1];
            c[2] = (__bf16)v[2]; c[3] = (__bf16)v[3];
            *(bf16x4*)&As[mm * LDST + kq] = c;
        }
        // ---- stage B tile (128 x 64) ----
        if (Bk == 1) {                       // k-contiguous (weights, K matrices)
#pragma unroll
            for (int i = 0; i < 8; ++i) {
                int q  = tid + i * 256;
                int nn = q >> 4, kq = (q & 15) * 4;
                const float* src = Bp + (long long)(n0 + nn) * Bn + (k0 + kq);
                floatx4 v = *(const floatx4*)src;
                if (pf) __builtin_prefetch(src + BK, 0, 3);
                bf16x4 c;
                c[0] = (__bf16)v[0]; c[1] = (__bf16)v[1];
                c[2] = (__bf16)v[2]; c[3] = (__bf16)v[3];
                *(bf16x4*)&Bs[nn * LDST + kq] = c;
            }
        } else if (Bn == 1) {                // n-contiguous (implicit-transpose V)
#pragma unroll
            for (int i = 0; i < 8; ++i) {
                int q  = tid + i * 256;
                int kk = q >> 5, nq = (q & 31) * 4;
                const float* src = Bp + (n0 + nq) + (long long)(k0 + kk) * Bk;
                floatx4 v = *(const floatx4*)src;
                if (pf) __builtin_prefetch(src + (long long)BK * Bk, 0, 3);
                Bs[(nq + 0) * LDST + kk] = (__bf16)v[0];
                Bs[(nq + 1) * LDST + kk] = (__bf16)v[1];
                Bs[(nq + 2) * LDST + kk] = (__bf16)v[2];
                Bs[(nq + 3) * LDST + kk] = (__bf16)v[3];
            }
        } else {                             // generic scalar fallback
#pragma unroll
            for (int i = 0; i < 32; ++i) {
                int e  = tid + i * 256;
                int nn = e >> 6, kk = e & 63;
                Bs[nn * LDST + kk] =
                    (__bf16)Bp[(long long)(n0 + nn) * Bn + (long long)(k0 + kk) * Bk];
            }
        }
        __syncthreads();

        // ---- two K=32 sub-steps per staged tile: 16 WMMAs per barrier pair ----
#pragma unroll
        for (int kc = 0; kc < 2; ++kc) {
            bf16x16 a[2], b[4];
#pragma unroll
            for (int mt = 0; mt < 2; ++mt) {
                const __bf16* p = &As[(wm * 32 + mt * 16 + lr) * LDST + kc * 32];
#pragma unroll
                for (int j = 0; j < 8; ++j) {
                    a[mt][j]     = p[g * 8 + j];        // K 0..7  / 8..15
                    a[mt][8 + j] = p[16 + g * 8 + j];   // K 16..23 / 24..31
                }
            }
#pragma unroll
            for (int nt = 0; nt < 4; ++nt) {
                const __bf16* p = &Bs[(wn * 64 + nt * 16 + lr) * LDST + kc * 32];
#pragma unroll
                for (int j = 0; j < 8; ++j) {
                    b[nt][j]     = p[g * 8 + j];        // K 0..7  / 8..15
                    b[nt][8 + j] = p[16 + g * 8 + j];   // K 16..23 / 24..31
                }
            }
#pragma unroll
            for (int mt = 0; mt < 2; ++mt)
#pragma unroll
                for (int nt = 0; nt < 4; ++nt)
                    acc[mt][nt] = __builtin_amdgcn_wmma_f32_16x16x32_bf16(
                        false, a[mt], false, b[nt], (short)0, acc[mt][nt], false, false);
        }
        __syncthreads();
    }

    // ---- store C (f32) per 16x16 C/D layout ----
#pragma unroll
    for (int mt = 0; mt < 2; ++mt)
#pragma unroll
        for (int nt = 0; nt < 4; ++nt) {
            int row0 = m0 + wm * 32 + mt * 16 + g * 8;
            int col  = n0 + wn * 64 + nt * 16 + lr;
#pragma unroll
            for (int i = 0; i < 8; ++i)
                Cp[(long long)(row0 + i) * Cm + col] = acc[mt][nt][i];
        }
}

// ---------------- embedding * sqrt(D) ----------------
__global__ __launch_bounds__(256) void embed_k(const int* __restrict__ ids,
                                               const float* __restrict__ emb,
                                               float* __restrict__ h)
{
    int t = blockIdx.x;
    long long id = ids[t];
    for (int d = threadIdx.x; d < D_; d += 256)
        h[(long long)t * D_ + d] = emb[id * D_ + d] * EMB_SCALE_;
}

// ---------------- RMSNorm: out = x * rsqrt(mean(x^2)+eps) * (1+w) ----------------
__global__ __launch_bounds__(256) void rmsnorm_k(const float* __restrict__ x,
                                                 const float* __restrict__ w,
                                                 float* __restrict__ out, int C)
{
    __shared__ float red[256];
    int tid = threadIdx.x;
    long long base = (long long)blockIdx.x * C;
    float s = 0.f;
    for (int c = tid; c < C; c += 256) { float v = x[base + c]; s += v * v; }
    red[tid] = s; __syncthreads();
    for (int st = 128; st > 0; st >>= 1) { if (tid < st) red[tid] += red[tid + st]; __syncthreads(); }
    float rs = rsqrtf(red[0] / (float)C + EPS_);
    for (int c = tid; c < C; c += 256)
        out[base + c] = x[base + c] * rs * (1.f + w[c]);
}

// ---------------- residual: h += rms(y, w) ----------------
__global__ __launch_bounds__(256) void rms_residual_k(float* __restrict__ h,
                                                      const float* __restrict__ y,
                                                      const float* __restrict__ w, int C)
{
    __shared__ float red[256];
    int tid = threadIdx.x;
    long long base = (long long)blockIdx.x * C;
    float s = 0.f;
    for (int c = tid; c < C; c += 256) { float v = y[base + c]; s += v * v; }
    red[tid] = s; __syncthreads();
    for (int st = 128; st > 0; st >>= 1) { if (tid < st) red[tid] += red[tid + st]; __syncthreads(); }
    float rs = rsqrtf(red[0] / (float)C + EPS_);
    for (int c = tid; c < C; c += 256)
        h[base + c] += y[base + c] * rs * (1.f + w[c]);
}

// ---------------- fused per-head RMSNorm + RoPE (HD=256 == blockDim) ----------------
__global__ __launch_bounds__(256) void rms_rope_k(float* __restrict__ buf,
                                                  const float* __restrict__ w,
                                                  const float* __restrict__ cosT,
                                                  const float* __restrict__ sinT,
                                                  const int* __restrict__ pos, int nheads)
{
    __shared__ float red[256];
    __shared__ float sh[HD_];
    int t = blockIdx.x, hh = blockIdx.y, d = threadIdx.x;
    float* x = buf + ((long long)t * nheads + hh) * HD_;
    float v = x[d];
    red[d] = v * v; __syncthreads();
    for (int st = 128; st > 0; st >>= 1) { if (d < st) red[d] += red[d + st]; __syncthreads(); }
    float rs = rsqrtf(red[0] / (float)HD_ + EPS_);
    float nv = v * rs * (1.f + w[d]);
    sh[d] = nv; __syncthreads();
    float rot = (d < HD_ / 2) ? -sh[d + HD_ / 2] : sh[d - HD_ / 2];
    long long p = pos[t];
    x[d] = nv * cosT[p * HD_ + d] + rot * sinT[p * HD_ + d];
}

// ---------------- causal scaled softmax over S[h][t][0..T); zero for c>t ----------------
__global__ __launch_bounds__(256) void softmax_causal_k(float* __restrict__ S)
{
    __shared__ float red[256];
    int t = blockIdx.x, h = blockIdx.y, tid = threadIdx.x;
    float* row = S + ((long long)h * T_ + t) * (long long)T_;
    float m = -1e30f;
    for (int c = tid; c < T_; c += 256)
        if (c <= t) m = fmaxf(m, row[c] * SCALE_);
    red[tid] = m; __syncthreads();
    for (int st = 128; st > 0; st >>= 1) { if (tid < st) red[tid] = fmaxf(red[tid], red[tid + st]); __syncthreads(); }
    m = red[0]; __syncthreads();
    float sum = 0.f;
    for (int c = tid; c < T_; c += 256) {
        float e = (c <= t) ? __expf(row[c] * SCALE_ - m) : 0.f;
        row[c] = e; sum += e;
    }
    red[tid] = sum; __syncthreads();
    for (int st = 128; st > 0; st >>= 1) { if (tid < st) red[tid] += red[tid + st]; __syncthreads(); }
    float inv = 1.f / red[0];
    for (int c = tid; c < T_; c += 256) row[c] *= inv;
}

// ---------------- gate = gelu_tanh(gate) * up ----------------
__global__ __launch_bounds__(256) void gelu_mul_k(float* __restrict__ gate,
                                                  const float* __restrict__ up,
                                                  long long n)
{
    long long i = (long long)blockIdx.x * 256 + threadIdx.x;
    if (i < n) {
        float x = gate[i];
        float inner = 0.7978845608028654f * (x + 0.044715f * x * x * x);
        gate[i] = 0.5f * x * (1.f + tanhf(inner)) * up[i];
    }
}

// ---------------- last-token logits: async-LDS staged GEMV + block argmax ----------------
__global__ __launch_bounds__(256) void logits_part_k(const float* __restrict__ xlast,
                                                     const float* __restrict__ lm,
                                                     float* __restrict__ pv,
                                                     int* __restrict__ pi)
{
    __shared__ float sx[D_];       // 8KB: last-token activations, reused by all rows
    __shared__ float sv[256];
    __shared__ int   si[256];
    int tid = threadIdx.x;

    // Stage xlast -> LDS via CDNA5 async global->LDS copy (32B per lane).
    {
        unsigned lds_off = (unsigned)(unsigned long long)(const void*)&sx[tid * 8];
        unsigned long long ga = (unsigned long long)(const void*)(xlast + tid * 8);
        asm volatile(
            "global_load_async_to_lds_b128 %0, %1, off\n\t"
            "global_load_async_to_lds_b128 %0, %1, off offset:16"
            :: "v"(lds_off), "v"(ga) : "memory");
        asm volatile("s_wait_asynccnt 0" ::: "memory");
    }
    __syncthreads();

    int v = blockIdx.x * 256 + tid;
    const float* r = lm + (long long)v * D_;
    float s = 0.f;
    for (int i = 0; i < D_; i += 4) {
        floatx4 rv = *(const floatx4*)(r + i);
        s += rv[0] * sx[i] + rv[1] * sx[i + 1] + rv[2] * sx[i + 2] + rv[3] * sx[i + 3];
    }
    sv[tid] = s; si[tid] = v; __syncthreads();
    for (int st = 128; st > 0; st >>= 1) {
        if (tid < st) {
            if (sv[tid + st] > sv[tid] ||
                (sv[tid + st] == sv[tid] && si[tid + st] < si[tid])) {
                sv[tid] = sv[tid + st]; si[tid] = si[tid + st];
            }
        }
        __syncthreads();
    }
    if (tid == 0) { pv[blockIdx.x] = sv[0]; pi[blockIdx.x] = si[0]; }
}

__global__ __launch_bounds__(256) void logits_final_k(const float* __restrict__ pv,
                                                      const int* __restrict__ pi,
                                                      float* __restrict__ out, int nb)
{
    __shared__ float sv[256];
    __shared__ int   si[256];
    int tid = threadIdx.x;
    sv[tid] = (tid < nb) ? pv[tid] : -1e30f;
    si[tid] = (tid < nb) ? pi[tid] : 0x7fffffff;
    __syncthreads();
    for (int st = 128; st > 0; st >>= 1) {
        if (tid < st) {
            if (sv[tid + st] > sv[tid] ||
                (sv[tid + st] == sv[tid] && si[tid + st] < si[tid])) {
                sv[tid] = sv[tid + st]; si[tid] = si[tid + st];
            }
        }
        __syncthreads();
    }
    if (tid == 0) { out[0] = (float)si[0]; out[1] = sv[0]; }
}

// ---------------- host orchestration ----------------
extern "C" void kernel_launch(void* const* d_in, const int* in_sizes, int n_in,
                              void* d_out, int out_size, void* d_ws, size_t ws_size,
                              hipStream_t stream)
{
    (void)in_sizes; (void)n_in; (void)out_size; (void)ws_size;

    const int*   input_ids   = (const int*)d_in[0];
    const int*   pos_ids     = (const int*)d_in[1];
    // d_in[2] causal_mask, d_in[3] update_mask: encoded analytically in kernels
    const float* embed       = (const float*)d_in[4];
    const float* ln_in       = (const float*)d_in[5];
    const float* q_w         = (const float*)d_in[6];
    const float* q_norm_w    = (const float*)d_in[7];
    const float* k_w         = (const float*)d_in[8];
    const float* k_norm_w    = (const float*)d_in[9];
    const float* v_w         = (const float*)d_in[10];
    const float* o_w         = (const float*)d_in[11];
    const float* ln_post_attn= (const float*)d_in[12];
    const float* ln_pre_ff   = (const float*)d_in[13];
    const float* gate_w      = (const float*)d_in[14];
    const float* up_w        = (const float*)d_in[15];
    const float* down_w      = (const float*)d_in[16];
    const float* ln_post_ff  = (const float*)d_in[17];
    const float* norm_w      = (const float*)d_in[18];
    const float* lm_head     = (const float*)d_in[19];
    // d_in[20] kv_cache (all zeros; absorbed into causal attention over T slots)
    const float* cos_s       = (const float*)d_in[21];
    const float* sin_s       = (const float*)d_in[22];
    const float* cos_f       = (const float*)d_in[23];
    const float* sin_f       = (const float*)d_in[24];

    // workspace carve-up (floats)
    float* ws = (float*)d_ws;
    size_t off = 0;
    auto alloc = [&](size_t n) { float* p = ws + off; off += n; return p; };
    float* hbuf  = alloc((size_t)T_ * D_);
    float* xb    = alloc((size_t)T_ * D_);
    float* qb    = alloc((size_t)T_ * H_ * HD_);
    float* kb    = alloc((size_t)T_ * KV_ * HD_);
    float* vb    = alloc((size_t)T_ * KV_ * HD_);
    float* Sb    = alloc((size_t)H_ * T_ * T_);
    float* ob    = alloc((size_t)T_ * H_ * HD_);
    float* yb    = alloc((size_t)T_ * D_);
    float* gateb = alloc((size_t)T_ * FF_);
    float* upb   = alloc((size_t)T_ * FF_);
    float* xlast = alloc((size_t)D_);
    float* pv    = alloc(128);
    int*   pi    = (int*)alloc(128);

    auto gemm = [&](const float* A, const float* B, float* C,
                    int M, int N, int K, int Am, int Bn, int Bk, int Cm,
                    long long Ab, int Abdiv, long long Bb, int Bbdiv, long long Cb,
                    int batch) {
        dim3 grid(N / BN, M / BM, batch);
        wmma_gemm_bf16<<<grid, 256, 0, stream>>>(A, B, C, M, N, K, Am, Bn, Bk, Cm,
                                                 Ab, Abdiv, Bb, Bbdiv, Cb);
    };

    // h = embed[ids] * sqrt(D)
    embed_k<<<T_, 256, 0, stream>>>(input_ids, embed, hbuf);

    for (int l = 0; l < L_; ++l) {
        const bool is_full = (l % 2 == 1);
        const float* cosT = is_full ? cos_f : cos_s;
        const float* sinT = is_full ? sin_f : sin_s;

        // x = rms(h, ln_in[l])
        rmsnorm_k<<<T_, 256, 0, stream>>>(hbuf, ln_in + (size_t)l * D_, xb, D_);

        // q,k,v projections
        gemm(xb, q_w + (size_t)l * H_ * HD_ * D_, qb,
             T_, H_ * HD_, D_, D_, D_, 1, H_ * HD_, 0, 1, 0, 1, 0, 1);
        gemm(xb, k_w + (size_t)l * KV_ * HD_ * D_, kb,
             T_, KV_ * HD_, D_, D_, D_, 1, KV_ * HD_, 0, 1, 0, 1, 0, 1);
        gemm(xb, v_w + (size_t)l * KV_ * HD_ * D_, vb,
             T_, KV_ * HD_, D_, D_, D_, 1, KV_ * HD_, 0, 1, 0, 1, 0, 1);

        // per-head RMS + RoPE
        { dim3 g(T_, H_);  rms_rope_k<<<g, HD_, 0, stream>>>(qb, q_norm_w + (size_t)l * HD_, cosT, sinT, pos_ids, H_); }
        { dim3 g(T_, KV_); rms_rope_k<<<g, HD_, 0, stream>>>(kb, k_norm_w + (size_t)l * HD_, cosT, sinT, pos_ids, KV_); }

        // S[h] = q_h @ k_h^T     (batch = H, kv head = h/NREP)
        gemm(qb, kb, Sb,
             T_, T_, HD_, H_ * HD_, KV_ * HD_, 1, T_,
             HD_, 1, HD_, NREP_, (long long)T_ * T_, H_);

        // causal softmax (scale + mask + normalize, zeros for c>t)
        { dim3 g(T_, H_); softmax_causal_k<<<g, 256, 0, stream>>>(Sb); }

        // O[h] = P[h] @ V_h       (B = V with implicit transpose: Bn=1, Bk=KV*HD)
        gemm(Sb, vb, ob,
             T_, HD_, T_, T_, 1, KV_ * HD_, H_ * HD_,
             (long long)T_ * T_, 1, HD_, NREP_, HD_, H_);

        // attn out projection
        gemm(ob, o_w + (size_t)l * D_ * H_ * HD_, yb,
             T_, D_, H_ * HD_, H_ * HD_, H_ * HD_, 1, D_, 0, 1, 0, 1, 0, 1);

        // h += rms(y, ln_post_attn[l])
        rms_residual_k<<<T_, 256, 0, stream>>>(hbuf, yb, ln_post_attn + (size_t)l * D_, D_);

        // x2 = rms(h, ln_pre_ff[l])
        rmsnorm_k<<<T_, 256, 0, stream>>>(hbuf, ln_pre_ff + (size_t)l * D_, xb, D_);

        // MLP
        gemm(xb, gate_w + (size_t)l * FF_ * D_, gateb,
             T_, FF_, D_, D_, D_, 1, FF_, 0, 1, 0, 1, 0, 1);
        gemm(xb, up_w + (size_t)l * FF_ * D_, upb,
             T_, FF_, D_, D_, D_, 1, FF_, 0, 1, 0, 1, 0, 1);

        long long n = (long long)T_ * FF_;
        gelu_mul_k<<<(unsigned)(n / 256), 256, 0, stream>>>(gateb, upb, n);

        gemm(gateb, down_w + (size_t)l * D_ * FF_, yb,
             T_, D_, FF_, FF_, FF_, 1, D_, 0, 1, 0, 1, 0, 1);

        // h += rms(mlp, ln_post_ff[l])
        rms_residual_k<<<T_, 256, 0, stream>>>(hbuf, yb, ln_post_ff + (size_t)l * D_, D_);
    }

    // final norm on last token only
    rmsnorm_k<<<1, 256, 0, stream>>>(hbuf + (size_t)(T_ - 1) * D_, norm_w, xlast, D_);

    // logits + argmax
    logits_part_k<<<V_ / 256, 256, 0, stream>>>(xlast, lm_head, pv, pi);
    logits_final_k<<<1, 256, 0, stream>>>(pv, pi, (float*)d_out, V_ / 256);
}
